// QNet_31507880083985
// MI455X (gfx1250) — matrix operand order repaired and checked by reading, not compiled, not run on hardware
//
#include <hip/hip_runtime.h>
#include <hip/hip_bf16.h>

typedef __attribute__((ext_vector_type(16))) _Float16 v16h;
typedef __attribute__((ext_vector_type(8)))  float    v8f;

union F16x16 { v16h v; _Float16 h[16]; uint4 q[2]; };
union F32x8  { v8f  v; float f[8]; };

#define WSTR 136  // padded LDS row stride (halves): 272B -> 16B-aligned b128, conflict-free

__device__ __forceinline__ v8f wmma_f16(v16h a, v16h b, v8f c) {
  return __builtin_amdgcn_wmma_f32_16x16x32_f16(false, a, false, b, (short)0, c, false, false);
}

// A-fragment (16x32 f16): this lane's global f32 row pointer rp, k-base kb.
// lanes 0-15: K = kb+0..15 (h0..15); lanes 16-31: K = kb+16..31.
__device__ __forceinline__ v16h loadA_g(const float* rp, int kb, int hi) {
  F16x16 f;
  const float4* p0 = (const float4*)(rp + kb + hi * 16);
  const float4* p1 = (const float4*)(rp + kb + 8 + hi * 16);
  float4 a = p0[0], b = p0[1], c = p1[0], d = p1[1];
  f.h[0]  = (_Float16)a.x; f.h[1]  = (_Float16)a.y; f.h[2]  = (_Float16)a.z; f.h[3]  = (_Float16)a.w;
  f.h[4]  = (_Float16)b.x; f.h[5]  = (_Float16)b.y; f.h[6]  = (_Float16)b.z; f.h[7]  = (_Float16)b.w;
  f.h[8]  = (_Float16)c.x; f.h[9]  = (_Float16)c.y; f.h[10] = (_Float16)c.z; f.h[11] = (_Float16)c.w;
  f.h[12] = (_Float16)d.x; f.h[13] = (_Float16)d.y; f.h[14] = (_Float16)d.z; f.h[15] = (_Float16)d.w;
  return f.v;
}

// A-fragment from LDS f16 row (row = &stg[l16*WSTR])
__device__ __forceinline__ v16h loadA_s(const _Float16* row, int kb, int hi) {
  F16x16 f;
  f.q[0] = *(const uint4*)(row + kb + hi * 16);
  f.q[1] = *(const uint4*)(row + kb + 8 + hi * 16);
  return f.v;
}

// B-fragment from LDS column-major weights: base = &w[col*WSTR + hi*16], 16 consecutive K
__device__ __forceinline__ v16h loadB_s(const _Float16* base, int kb) {
  F16x16 f;
  f.q[0] = *(const uint4*)(base + kb);
  f.q[1] = *(const uint4*)(base + kb + 8);
  return f.v;
}

// B-fragment gathered from row-major [32][96] f32 gate weights (hoisted, cached in WGP$)
__device__ __forceinline__ v16h loadB_gate(const float* w, int col, int hi) {
  F16x16 f;
#pragma unroll
  for (int k = 0; k < 16; ++k) f.h[k] = (_Float16)w[(hi * 16 + k) * 96 + col];
  return f.v;
}

__device__ __forceinline__ float sigm(float x) { return 1.f / (1.f + __expf(-x)); }

// ---------------- comm path: dominant GEMM chain -> comm_values [B,N,32] ----------------
__global__ void __launch_bounds__(128) comm_path_kernel(
    const float* __restrict__ comm, const float* __restrict__ hidden,
    const float* __restrict__ Wc1, const float* __restrict__ bc1,
    const float* __restrict__ Wc2, const float* __restrict__ bc2,
    const float* __restrict__ Wihc, const float* __restrict__ Whhc,
    const float* __restrict__ bihc, const float* __restrict__ bhhc,
    float* __restrict__ comm_values)
{
  __shared__ __align__(16) _Float16 w1T[128 * WSTR];  // 34816 B
  __shared__ __align__(16) _Float16 w2T[32 * WSTR];   //  8704 B
  __shared__ __align__(16) _Float16 stg[4][16 * WSTR];// 17408 B  (total 60928 B)

  const int ag = blockIdx.y;
  for (int e = threadIdx.x; e < 128 * 128; e += 128) {
    int k = e >> 7, c = e & 127;
    w1T[c * WSTR + k] = (_Float16)Wc1[(((size_t)ag * 128) + k) * 128 + c];
  }
  for (int e = threadIdx.x; e < 128 * 32; e += 128) {
    int k = e >> 5, c = e & 31;
    w2T[c * WSTR + k] = (_Float16)Wc2[(((size_t)ag * 128) + k) * 32 + c];
  }
  __syncthreads();

  const int lane = threadIdx.x & 31, wv = threadIdx.x >> 5;
  const int l16 = lane & 15, hi = lane >> 4;
  _Float16* myst = stg[wv];

  float b1v[8], b2v[2], bgv[4], bxnv[2], bhnv[2];
#pragma unroll
  for (int t = 0; t < 8; ++t) b1v[t] = bc1[ag * 128 + t * 16 + l16];
#pragma unroll
  for (int t = 0; t < 2; ++t) b2v[t] = bc2[ag * 32 + t * 16 + l16];
#pragma unroll
  for (int t = 0; t < 4; ++t)
    bgv[t] = bihc[ag * 96 + t * 16 + l16] + bhhc[ag * 96 + t * 16 + l16];
#pragma unroll
  for (int t = 0; t < 2; ++t) {
    bxnv[t] = bihc[ag * 96 + 64 + t * 16 + l16];
    bhnv[t] = bhhc[ag * 96 + 64 + t * 16 + l16];
  }
  // hoist the 12 tile-invariant gate B-fragments into registers (cols t*16+l16 cover 0..95)
  const float* wih = Wihc + ag * 32 * 96;
  const float* whh = Whhc + ag * 32 * 96;
  v16h gbx[6], gbh[6];
#pragma unroll
  for (int t = 0; t < 6; ++t) {
    gbx[t] = loadB_gate(wih, t * 16 + l16, hi);
    gbh[t] = loadB_gate(whh, t * 16 + l16, hi);
  }
  const v8f z8 = {0.f, 0.f, 0.f, 0.f, 0.f, 0.f, 0.f, 0.f};

  for (int tile = 0; tile < 8; ++tile) {
    const int bb = (blockIdx.x * 4 + wv) * 16 + tile * 2;  // 2 batches per 16-row tile
    const int rb = bb + (l16 >> 3);                        // this lane's row -> (b, j)
    const int rj = l16 & 7;
    const float* rp = comm + ((((size_t)rb * 8 + ag) * 8 + rj) << 7);

    // ---- L1: relu(A[16x128] x Wc1 + bc1) ----
    F32x8 acc1[8];
#pragma unroll
    for (int t = 0; t < 8; ++t) acc1[t].v = z8;
#pragma unroll
    for (int kk = 0; kk < 4; ++kk) {
      v16h a = loadA_g(rp, kk * 32, hi);
#pragma unroll
      for (int t = 0; t < 8; ++t) {
        v16h b = loadB_s(&w1T[(t * 16 + l16) * WSTR + hi * 16], kk * 32);
        acc1[t].v = wmma_f16(a, b, acc1[t].v);
      }
    }
#pragma unroll
    for (int t = 0; t < 8; ++t) {
#pragma unroll
      for (int r = 0; r < 8; ++r) {
        float v = fmaxf(acc1[t].f[r] + b1v[t], 0.f);
        myst[(r + hi * 8) * WSTR + t * 16 + l16] = (_Float16)v;   // restage as A-source
      }
    }
    // ---- L2: relu(X1 x Wc2 + bc2) ----
    F32x8 acc2[2];
#pragma unroll
    for (int t = 0; t < 2; ++t) acc2[t].v = z8;
#pragma unroll
    for (int kk = 0; kk < 4; ++kk) {
      v16h a = loadA_s(&myst[l16 * WSTR], kk * 32, hi);
#pragma unroll
      for (int t = 0; t < 2; ++t) {
        v16h b = loadB_s(&w2T[(t * 16 + l16) * WSTR + hi * 16], kk * 32);
        acc2[t].v = wmma_f16(a, b, acc2[t].v);
      }
    }
#pragma unroll
    for (int t = 0; t < 2; ++t) {
#pragma unroll
      for (int r = 0; r < 8; ++r) {
        float v = fmaxf(acc2[t].f[r] + b2v[t], 0.f);
        myst[(r + hi * 8) * WSTR + t * 16 + l16] = (_Float16)v;   // X2 16x32
      }
    }
    // ---- gates: gxc = X2 x Wihc, ghc = hidden(b(row)) x Whhc (K=32, one frag) ----
    v16h ax = loadA_s(&myst[l16 * WSTR], 0, hi);
    const float* hp = hidden + (((size_t)rb * 8 + ag) << 5);
    v16h ah = loadA_g(hp, 0, hi);

    F32x8 gacc[4], xn[2], hn[2];
#pragma unroll
    for (int t = 0; t < 4; ++t) {                 // r,z gates: fused via C-chaining
      v8f tmp = wmma_f16(ah, gbh[t], z8);
      gacc[t].v = wmma_f16(ax, gbx[t], tmp);
    }
#pragma unroll
    for (int t = 0; t < 2; ++t) {                 // n gate: xn / hn kept separate
      xn[t].v = wmma_f16(ax, gbx[4 + t], z8);
      hn[t].v = wmma_f16(ah, gbh[4 + t], z8);
    }
    // ---- GRU + masked mean over j (j == accumulator VGPR index r) ----
#pragma unroll
    for (int t2 = 0; t2 < 2; ++t2) {
      const int bH = bb + hi;                     // lanes 0-15 -> b0, 16-31 -> b1
      float h = hidden[(((size_t)bH * 8 + ag) << 5) + t2 * 16 + l16];
      float sum = 0.f;
#pragma unroll
      for (int r = 0; r < 8; ++r) {
        float xr = gacc[t2].f[r] + bgv[t2];
        float xz = gacc[2 + t2].f[r] + bgv[2 + t2];
        float xnn = xn[t2].f[r] + bxnv[t2];
        float hnn = hn[t2].f[r] + bhnv[t2];
        float rg = sigm(xr);
        float zg = sigm(xz);
        float ng = tanhf(xnn + rg * hnn);
        float hc = (1.f - zg) * ng + zg * h;
        sum += (r == ag) ? 0.f : hc;
      }
      comm_values[(((size_t)bH * 8 + ag) << 5) + t2 * 16 + l16] = sum * (1.f / 7.f);
    }
  }
}

// ---------------- own path: obs -> MLP -> GRU -> next_h [B,N,32] ----------------
__global__ void __launch_bounds__(128) own_path_kernel(
    const float* __restrict__ obs, const float* __restrict__ hidden,
    const float* __restrict__ W1, const float* __restrict__ b1,
    const float* __restrict__ W2, const float* __restrict__ b2,
    const float* __restrict__ Wih, const float* __restrict__ Whh,
    const float* __restrict__ bih, const float* __restrict__ bhh,
    float* __restrict__ next_h)
{
  __shared__ __align__(16) _Float16 w1T[128 * WSTR];
  __shared__ __align__(16) _Float16 w2T[32 * WSTR];
  __shared__ __align__(16) _Float16 stg[4][16 * WSTR];

  const int ag = blockIdx.y;
  for (int e = threadIdx.x; e < 128 * 128; e += 128) {
    int k = e >> 7, c = e & 127;
    w1T[c * WSTR + k] = (_Float16)W1[(((size_t)ag * 128) + k) * 128 + c];
  }
  for (int e = threadIdx.x; e < 128 * 32; e += 128) {
    int k = e >> 5, c = e & 31;
    w2T[c * WSTR + k] = (_Float16)W2[(((size_t)ag * 128) + k) * 32 + c];
  }
  __syncthreads();

  const int lane = threadIdx.x & 31, wv = threadIdx.x >> 5;
  const int l16 = lane & 15, hi = lane >> 4;
  _Float16* myst = stg[wv];

  float b1v[8], b2v[2], bgv[4], bxnv[2], bhnv[2];
#pragma unroll
  for (int t = 0; t < 8; ++t) b1v[t] = b1[ag * 128 + t * 16 + l16];
#pragma unroll
  for (int t = 0; t < 2; ++t) b2v[t] = b2[ag * 32 + t * 16 + l16];
#pragma unroll
  for (int t = 0; t < 4; ++t)
    bgv[t] = bih[ag * 96 + t * 16 + l16] + bhh[ag * 96 + t * 16 + l16];
#pragma unroll
  for (int t = 0; t < 2; ++t) {
    bxnv[t] = bih[ag * 96 + 64 + t * 16 + l16];
    bhnv[t] = bhh[ag * 96 + 64 + t * 16 + l16];
  }
  const float* wihp = Wih + ag * 32 * 96;
  const float* whhp = Whh + ag * 32 * 96;
  v16h gbx[6], gbh[6];
#pragma unroll
  for (int t = 0; t < 6; ++t) {
    gbx[t] = loadB_gate(wihp, t * 16 + l16, hi);
    gbh[t] = loadB_gate(whhp, t * 16 + l16, hi);
  }
  const v8f z8 = {0.f, 0.f, 0.f, 0.f, 0.f, 0.f, 0.f, 0.f};

  for (int tile = 0; tile < 4; ++tile) {
    const int bb = (blockIdx.x * 4 + wv) * 64 + tile * 16;  // 16 batches per tile
    const int rb = bb + l16;
    const float* rp = obs + (((size_t)rb * 8 + ag) << 7);

    F32x8 acc1[8];
#pragma unroll
    for (int t = 0; t < 8; ++t) acc1[t].v = z8;
#pragma unroll
    for (int kk = 0; kk < 4; ++kk) {
      v16h a = loadA_g(rp, kk * 32, hi);
#pragma unroll
      for (int t = 0; t < 8; ++t) {
        v16h b = loadB_s(&w1T[(t * 16 + l16) * WSTR + hi * 16], kk * 32);
        acc1[t].v = wmma_f16(a, b, acc1[t].v);
      }
    }
#pragma unroll
    for (int t = 0; t < 8; ++t) {
#pragma unroll
      for (int r = 0; r < 8; ++r) {
        float v = fmaxf(acc1[t].f[r] + b1v[t], 0.f);
        myst[(r + hi * 8) * WSTR + t * 16 + l16] = (_Float16)v;
      }
    }
    F32x8 acc2[2];
#pragma unroll
    for (int t = 0; t < 2; ++t) acc2[t].v = z8;
#pragma unroll
    for (int kk = 0; kk < 4; ++kk) {
      v16h a = loadA_s(&myst[l16 * WSTR], kk * 32, hi);
#pragma unroll
      for (int t = 0; t < 2; ++t) {
        v16h b = loadB_s(&w2T[(t * 16 + l16) * WSTR + hi * 16], kk * 32);
        acc2[t].v = wmma_f16(a, b, acc2[t].v);
      }
    }
#pragma unroll
    for (int t = 0; t < 2; ++t) {
#pragma unroll
      for (int r = 0; r < 8; ++r) {
        float v = fmaxf(acc2[t].f[r] + b2v[t], 0.f);
        myst[(r + hi * 8) * WSTR + t * 16 + l16] = (_Float16)v;
      }
    }
    v16h ax = loadA_s(&myst[l16 * WSTR], 0, hi);
    const float* hp = hidden + (((size_t)rb * 8 + ag) << 5);
    v16h ah = loadA_g(hp, 0, hi);

    F32x8 gacc[4], xn[2], hn[2];
#pragma unroll
    for (int t = 0; t < 4; ++t) {
      v8f tmp = wmma_f16(ah, gbh[t], z8);
      gacc[t].v = wmma_f16(ax, gbx[t], tmp);
    }
#pragma unroll
    for (int t = 0; t < 2; ++t) {
      xn[t].v = wmma_f16(ax, gbx[4 + t], z8);
      hn[t].v = wmma_f16(ah, gbh[4 + t], z8);
    }
#pragma unroll
    for (int t2 = 0; t2 < 2; ++t2) {
#pragma unroll
      for (int r = 0; r < 8; ++r) {
        const int bR = bb + r + hi * 8;
        size_t off = (((size_t)bR * 8 + ag) << 5) + t2 * 16 + l16;
        float h = hidden[off];
        float xr = gacc[t2].f[r] + bgv[t2];
        float xz = gacc[2 + t2].f[r] + bgv[2 + t2];
        float xnn = xn[t2].f[r] + bxnv[t2];
        float hnn = hn[t2].f[r] + bhnv[t2];
        float rg = sigm(xr);
        float zg = sigm(xz);
        float ng = tanhf(xnn + rg * hnn);
        next_h[off] = (1.f - zg) * ng + zg * h;
      }
    }
  }
}

// ---------------- Q head: [next_h | comm_values] x Wq + bq ----------------
__global__ void __launch_bounds__(256) q_head_kernel(
    const float* __restrict__ next_h, const float* __restrict__ comm_values,
    const float* __restrict__ Wq, const float* __restrict__ bq,
    float* __restrict__ q)
{
  int idx = blockIdx.x * 256 + threadIdx.x;   // over B*N = 65536
  if (idx >= 8192 * 8) return;
  int n = idx & 7;
  const float* w = Wq + n * 64 * 14;
  float acc[14];
#pragma unroll
  for (int a = 0; a < 14; ++a) acc[a] = bq[n * 14 + a];
  const float* f1 = next_h + (size_t)idx * 32;
  const float* f2 = comm_values + (size_t)idx * 32;
  for (int k = 0; k < 32; ++k) {
    float x = f1[k];
#pragma unroll
    for (int a = 0; a < 14; ++a) acc[a] = fmaf(x, w[k * 14 + a], acc[a]);
  }
  for (int k = 0; k < 32; ++k) {
    float x = f2[k];
#pragma unroll
    for (int a = 0; a < 14; ++a) acc[a] = fmaf(x, w[(32 + k) * 14 + a], acc[a]);
  }
#pragma unroll
  for (int a = 0; a < 14; ++a) q[(size_t)idx * 14 + a] = acc[a];
}

extern "C" void kernel_launch(void* const* d_in, const int* in_sizes, int n_in,
                              void* d_out, int out_size, void* d_ws, size_t ws_size,
                              hipStream_t stream) {
  const float* obs    = (const float*)d_in[0];
  const float* hidden = (const float*)d_in[1];
  const float* comm   = (const float*)d_in[2];
  const float* W1  = (const float*)d_in[3];  const float* b1  = (const float*)d_in[4];
  const float* W2  = (const float*)d_in[5];  const float* b2  = (const float*)d_in[6];
  const float* Wc1 = (const float*)d_in[7];  const float* bc1 = (const float*)d_in[8];
  const float* Wc2 = (const float*)d_in[9];  const float* bc2 = (const float*)d_in[10];
  const float* Wih = (const float*)d_in[11]; const float* Whh = (const float*)d_in[12];
  const float* bih = (const float*)d_in[13]; const float* bhh = (const float*)d_in[14];
  const float* Wihc= (const float*)d_in[15]; const float* Whhc= (const float*)d_in[16];
  const float* bihc= (const float*)d_in[17]; const float* bhhc= (const float*)d_in[18];
  const float* Wq  = (const float*)d_in[19]; const float* bq  = (const float*)d_in[20];

  float* out = (float*)d_out;
  float* q      = out;                              // [B,N,14]
  float* next_h = out + (size_t)8192 * 8 * 14;      // [B,N,32]
  float* comm_values = (float*)d_ws;                // [B,N,32] scratch

  own_path_kernel<<<dim3(32, 8), 128, 0, stream>>>(
      obs, hidden, W1, b1, W2, b2, Wih, Whh, bih, bhh, next_h);
  comm_path_kernel<<<dim3(128, 8), 128, 0, stream>>>(
      comm, hidden, Wc1, bc1, Wc2, bc2, Wihc, Whhc, bihc, bhhc, comm_values);
  q_head_kernel<<<(8192 * 8 + 255) / 256, 256, 0, stream>>>(
      next_h, comm_values, Wq, bq, q);
}
